// CMADE_82995948027952
// MI455X (gfx1250) — matrix operand order repaired and checked by reading, not compiled, not run on hardware
//
#include <hip/hip_runtime.h>

// ---------------------------------------------------------------------------
// CMADE on gfx1250, v2:
//   pass 1: MW[b][j][k] = f16( W[j][k] * M[b][k][j] )      (LDS-transpose kernel)
//   pass 2: per-layer GEMM Y[b] = relu(X[b] @ MW[b].T + bias) with
//           V_WMMA_F32_16X16X32_F16 and double-buffered
//           GLOBAL_LOAD_ASYNC_TO_LDS_B128 staging (ASYNCcnt pipeline).
//   final : out = mean_b(X[b] @ MW3[b].T) + bias  (f32, deterministic in-block)
// Block = 128 threads = 4 wave32; tile 64x64; K-step 32.
// ---------------------------------------------------------------------------

typedef _Float16 h8  __attribute__((ext_vector_type(8)));
typedef _Float16 h16 __attribute__((ext_vector_type(16)));
typedef float    f8  __attribute__((ext_vector_type(8)));

#define NB   4096     // batch N
#define BMS  16       // ensemble size B
#define TPB  128      // threads per block (4 waves)
#define LDA  40       // padded row stride (halves) for As/Bs tiles (80B rows)

struct Smem {
  __align__(16) _Float16 As[2][64 * LDA];   // double-buffered A tile [n][k]
  __align__(16) _Float16 Bs[2][64 * LDA];   // double-buffered MW tile [j][k]
};

__device__ __forceinline__ uint32_t lds_addr(const void* p) {
  // generic pointer to LDS: low 32 bits are the LDS byte offset
  return (uint32_t)(uintptr_t)p;
}

// Stage one 64x32-half tile pair (A from Xb, B from MWb) asynchronously.
// Per thread: 4x global_load_async_to_lds_b128 (GVS addressing).
__device__ __forceinline__
void issue_tile(uint32_t ldsA, uint32_t ldsB,
                const _Float16* __restrict__ Xb, const _Float16* __restrict__ Wb,
                int K, int n0, int j0, int k0, int t)
{
  const int row = t >> 2;          // 0..31
  const int cc  = t & 3;           // 16B chunk within a 64B row
  const int a0 = ((n0 + row)      * K + k0 + cc * 8) * 2;   // byte offsets
  const int a1 = ((n0 + row + 32) * K + k0 + cc * 8) * 2;
  const int b0 = ((j0 + row)      * K + k0 + cc * 8) * 2;
  const int b1 = ((j0 + row + 32) * K + k0 + cc * 8) * 2;
  const uint32_t l0 = ldsA + row * 80 + cc * 16;
  const uint32_t l1 = ldsA + (row + 32) * 80 + cc * 16;
  const uint32_t l2 = ldsB + row * 80 + cc * 16;
  const uint32_t l3 = ldsB + (row + 32) * 80 + cc * 16;
  asm volatile(
      "global_load_async_to_lds_b128 %0, %4, %8\n\t"
      "global_load_async_to_lds_b128 %1, %5, %8\n\t"
      "global_load_async_to_lds_b128 %2, %6, %9\n\t"
      "global_load_async_to_lds_b128 %3, %7, %9"
      :: "v"(l0), "v"(l1), "v"(l2), "v"(l3),
         "v"(a0), "v"(a1), "v"(b0), "v"(b1),
         "s"(Xb), "s"(Wb)
      : "memory");
}

__device__ __forceinline__ void wait_async0() {
  asm volatile("s_wait_asynccnt 0x0" ::: "memory");
}

// Fragment gather + 4 WMMAs for one staged tile pair.
__device__ __forceinline__
void wmma_tile(const _Float16* __restrict__ As, const _Float16* __restrict__ Bs,
               int wave, int r, int hi, f8 acc[4])
{
  // B (32x16): lane<16 holds K=0..7 & 16..23, lane>=16 holds K=8..15 & 24..31, col = lane&15
  h8 bl = *(const h8*)(Bs + (wave * 16 + r) * LDA + hi * 8);
  h8 bh = *(const h8*)(Bs + (wave * 16 + r) * LDA + hi * 8 + 16);
  h16 bf = __builtin_shufflevector(bl, bh, 0,1,2,3,4,5,6,7,8,9,10,11,12,13,14,15);
#pragma unroll
  for (int mt = 0; mt < 4; ++mt) {
    // A (16x32): lane<16 holds K=0..7 & 16..23, lane>=16 holds K=8..15 & 24..31
    h8 al = *(const h8*)(As + (mt * 16 + r) * LDA + hi * 8);
    h8 ah = *(const h8*)(As + (mt * 16 + r) * LDA + hi * 8 + 16);
    h16 af = __builtin_shufflevector(al, ah, 0,1,2,3,4,5,6,7,8,9,10,11,12,13,14,15);
    acc[mt] = __builtin_amdgcn_wmma_f32_16x16x32_f16(
        false, af, false, bf, (short)0, acc[mt], false, false);
  }
}

// xy f32 -> f16 staging
__global__ __launch_bounds__(TPB)
void cmade_prep(const float* __restrict__ x, _Float16* __restrict__ y, int n) {
  int i = blockIdx.x * TPB + threadIdx.x;
  if (i < n) y[i] = (_Float16)x[i];
}

// Masked-weight precompute: MW[b][j][k] = f16(W[j][k] * M[b][k][j]).
// Mask read coalesced along j, transposed through LDS; W read + MW write
// coalesced along k. One pass over all mask elements (no per-n-block redo).
__global__ __launch_bounds__(256)
void cmade_maskw(const float* __restrict__ W, const float* __restrict__ Msk,
                 _Float16* __restrict__ MW, int K, int OUT)
{
  __shared__ float Ms[32][65];
  const int t  = threadIdx.x;
  const int k0 = blockIdx.x * 32;
  const int j0 = blockIdx.y * 64;
  const int b  = blockIdx.z;
  const float* Mb = Msk + (size_t)b * K * OUT;
#pragma unroll
  for (int it = 0; it < 8; ++it) {
    int e = it * 256 + t;          // 0..2047, j fastest -> coalesced
    int kk = e >> 6, jj = e & 63;
    Ms[kk][jj] = Mb[(size_t)(k0 + kk) * OUT + j0 + jj];
  }
  __syncthreads();
  _Float16* MWb = MW + (size_t)b * OUT * K;
#pragma unroll
  for (int it = 0; it < 8; ++it) {
    int e = it * 256 + t;          // k fastest -> coalesced W read, MW write
    int jj = e >> 5, kk = e & 31;
    float w = W[(size_t)(j0 + jj) * K + k0 + kk];
    MWb[(size_t)(j0 + jj) * K + k0 + kk] = (_Float16)(w * Ms[kk][jj]);
  }
}

// Hidden layers: Y[b] = relu(X[b] @ MW[b].T + bias), f16 out.
__global__ __launch_bounds__(TPB)
void cmade_layer(const _Float16* __restrict__ X, const _Float16* __restrict__ MW,
                 const float* __restrict__ bias, _Float16* __restrict__ Y,
                 int K, int OUT, long xbstride)
{
  __shared__ Smem sm;
  const int t    = threadIdx.x;
  const int b    = blockIdx.z;
  const int n0   = blockIdx.x * 64;
  const int j0   = blockIdx.y * 64;
  const int wave = t >> 5, lane = t & 31, r = lane & 15, hi = lane >> 4;

  const _Float16* Xb = X + (size_t)b * (size_t)xbstride;   // 0: broadcast xy
  const _Float16* Wb = MW + (size_t)b * (size_t)OUT * K;
  const uint32_t aB0 = lds_addr(&sm.As[0][0]), aB1 = lds_addr(&sm.As[1][0]);
  const uint32_t bB0 = lds_addr(&sm.Bs[0][0]), bB1 = lds_addr(&sm.Bs[1][0]);

  f8 acc[4] = {};
  issue_tile(aB0, bB0, Xb, Wb, K, n0, j0, 0, t);
  int p = 0;
  for (int k0 = 0; k0 < K; k0 += 32, p ^= 1) {
    wait_async0();        // this wave's stage of buffer p landed
    __syncthreads();      // all waves staged p; all reads of p^1 finished
    if (k0 + 32 < K)      // overlap next stage with this tile's WMMAs
      issue_tile(p ? aB0 : aB1, p ? bB0 : bB1, Xb, Wb, K, n0, j0, k0 + 32, t);
    wmma_tile(sm.As[p], sm.Bs[p], wave, r, hi, acc);
  }

  // C/D layout: col = lane&15, row = vgpr + 8*(lane>=16)
  const int   gcol = j0 + wave * 16 + r;
  const float bv   = bias[gcol];
  _Float16* Yb = Y + (size_t)b * (size_t)NB * OUT;
#pragma unroll
  for (int mt = 0; mt < 4; ++mt)
#pragma unroll
    for (int v = 0; v < 8; ++v) {
      const int grow = n0 + mt * 16 + hi * 8 + v;
      float val = acc[mt][v] + bv;
      Yb[(size_t)grow * OUT + gcol] = (_Float16)(val > 0.f ? val : 0.f);
    }
}

// Final layer: out = (1/B) * sum_b(X[b] @ MW3[b].T) + bias; flat b*k pipeline.
__global__ __launch_bounds__(TPB)
void cmade_final(const _Float16* __restrict__ X, const _Float16* __restrict__ MW,
                 const float* __restrict__ bias, float* __restrict__ out)
{
  __shared__ Smem sm;
  const int t    = threadIdx.x;
  const int n0   = blockIdx.x * 64;
  const int wave = t >> 5, lane = t & 31, r = lane & 15, hi = lane >> 4;
  const uint32_t aB0 = lds_addr(&sm.As[0][0]), aB1 = lds_addr(&sm.As[1][0]);
  const uint32_t bB0 = lds_addr(&sm.Bs[0][0]), bB1 = lds_addr(&sm.Bs[1][0]);

  const int STEPS = BMS * (1024 / 32);   // 512 K-steps across the ensemble
  f8 acc[4] = {};
  issue_tile(aB0, bB0, X, MW, 1024, n0, 0, 0, t);
  int p = 0;
  for (int s = 0; s < STEPS; ++s, p ^= 1) {
    wait_async0();
    __syncthreads();
    if (s + 1 < STEPS) {
      const int nb = (s + 1) >> 5;
      const int nk = ((s + 1) & 31) * 32;
      issue_tile(p ? aB0 : aB1, p ? bB0 : bB1,
                 X + (size_t)nb * NB * 1024, MW + (size_t)nb * 64 * 1024,
                 1024, n0, 0, nk, t);
    }
    wmma_tile(sm.As[p], sm.Bs[p], wave, r, hi, acc);
  }

  const int   gcol = wave * 16 + r;
  const float bv   = bias[gcol];
#pragma unroll
  for (int mt = 0; mt < 4; ++mt)
#pragma unroll
    for (int v = 0; v < 8; ++v) {
      const int grow = n0 + mt * 16 + hi * 8 + v;
      out[(size_t)grow * 64 + gcol] = acc[mt][v] * (1.0f / BMS) + bv;
    }
}

extern "C" void kernel_launch(void* const* d_in, const int* in_sizes, int n_in,
                              void* d_out, int out_size, void* d_ws, size_t ws_size,
                              hipStream_t stream) {
  (void)in_sizes; (void)n_in; (void)out_size; (void)ws_size;
  const float* xy = (const float*)d_in[0];
  const float* W0 = (const float*)d_in[1];
  const float* b0 = (const float*)d_in[2];
  const float* W1 = (const float*)d_in[3];
  const float* b1 = (const float*)d_in[4];
  const float* W2 = (const float*)d_in[5];
  const float* b2 = (const float*)d_in[6];
  const float* W3 = (const float*)d_in[7];
  const float* b3 = (const float*)d_in[8];
  const float* m0 = (const float*)d_in[9];
  const float* m1 = (const float*)d_in[10];
  const float* m2 = (const float*)d_in[11];
  const float* m3 = (const float*)d_in[12];
  float* out = (float*)d_out;

  // Workspace map (1MB-aligned regions): X0 | MW0..MW3 | A ping | A pong
  char* ws = (char*)d_ws;
  size_t off = 0;
  _Float16* X0  = (_Float16*)(ws + off); off += (size_t)1 << 20;              // 0.75MB used
  _Float16* MW0 = (_Float16*)(ws + off); off += (size_t)BMS * 1024 * 96 * 2;  // 3MB
  _Float16* MW1 = (_Float16*)(ws + off); off += (size_t)BMS * 1024 * 1024 * 2;// 32MB
  _Float16* MW2 = (_Float16*)(ws + off); off += (size_t)BMS * 1024 * 1024 * 2;// 32MB
  _Float16* MW3 = (_Float16*)(ws + off); off += (size_t)BMS * 64 * 1024 * 2;  // 2MB
  const size_t actBytes = (size_t)BMS * NB * 1024 * 2;                        // 128MB
  _Float16* Aping = (_Float16*)(ws + off); off += actBytes;
  _Float16* Apong = (_Float16*)(ws + off);

  dim3 blk(TPB);
  cmade_prep<<<(NB * 96 + TPB - 1) / TPB, blk, 0, stream>>>(xy, X0, NB * 96);

  // masked-weight precompute (one pass per layer)
  cmade_maskw<<<dim3(96 / 32,   1024 / 64, BMS), dim3(256), 0, stream>>>(W0, m0, MW0, 96,   1024);
  cmade_maskw<<<dim3(1024 / 32, 1024 / 64, BMS), dim3(256), 0, stream>>>(W1, m1, MW1, 1024, 1024);
  cmade_maskw<<<dim3(1024 / 32, 1024 / 64, BMS), dim3(256), 0, stream>>>(W2, m2, MW2, 1024, 1024);
  cmade_maskw<<<dim3(1024 / 32, 64 / 64,   BMS), dim3(256), 0, stream>>>(W3, m3, MW3, 1024, 64);

  dim3 g(NB / 64, 1024 / 64, BMS);
  cmade_layer<<<g, blk, 0, stream>>>(X0,    MW0, b0, Aping, 96,   1024, 0);
  cmade_layer<<<g, blk, 0, stream>>>(Aping, MW1, b1, Apong, 1024, 1024, (long)NB * 1024);
  cmade_layer<<<g, blk, 0, stream>>>(Apong, MW2, b2, Aping, 1024, 1024, (long)NB * 1024);
  cmade_final<<<dim3(NB / 64, 1, 1), blk, 0, stream>>>(Aping, MW3, b3, out);
}
